// S2S_17025250361721
// MI455X (gfx1250) — compile-verified, matrix operand order checked by reading, not run on hardware
//
#include <hip/hip_runtime.h>
#include <hip/hip_bf16.h>
#include <math.h>

// ---------------------------------------------------------------------------
// Seq2seq tanh-RNN forward for MI455X (gfx1250, wave32, WMMA).
//   B=16, T=256, D_CHAR=128, D_HID=512, V=32000
// Strategy:
//   * Convert weights + gathered embeddings to bf16 once (L2-resident).
//   * Input projections + giant output projection via v_wmma_f32_16x16x32_bf16.
//     Output GEMM uses K-outer / 5-way-N-inner blocking so each A fragment
//     feeds 5 WMMAs (better wmma:vmem ratio; B streams from the 192MB L2).
//   * Sequential recurrences run as ONE persistent workgroup (16 waves) with
//     the 16x512 hidden state double-buffered in LDS; Whh streamed from L2;
//     A fragment shared across both of a wave's N tiles per K-step.
//   * log-softmax: stage each 125KB row in dynamic LDS (320KB/WGP) -> exactly
//     one HBM read + one HBM write per logit.
// ---------------------------------------------------------------------------

typedef __bf16 bf16_t;
typedef __attribute__((ext_vector_type(16))) __bf16 v16bf;
typedef __attribute__((ext_vector_type(8)))  float  v8f;

#define B_SZ   16
#define T_SZ   256
#define DC     128
#define DH     512
#define V_SZ   32000
#define M_ROWS (B_SZ * T_SZ)   // 4096
#define NT_PER_WAVE 5          // out_gemm: 8 waves * 5 tiles * 16 = 640 cols/block

// ---- WMMA helpers ---------------------------------------------------------

__device__ __forceinline__ v8f wmma_bf16(v16bf a, v16bf b, v8f c) {
    // D = A(16x32 bf16) * B(32x16 bf16) + C(16x16 f32)
    return __builtin_amdgcn_wmma_f32_16x16x32_bf16(
        /*neg_a=*/false, a, /*neg_b=*/false, b,
        /*c_mod=*/(short)0, c, /*reuse_a=*/false, /*reuse_b=*/false);
}

// Load a 16x32 bf16 fragment (A-layout; B uses the symmetric layout with the
// row index playing the role of N).  lane = m + 16*half; per-lane element e:
//   k = e + 8*half          for e in [0,8)
//   k = e + 8 + 8*half      for e in [8,16)
// -> two contiguous 16-byte chunks per lane.
__device__ __forceinline__ v16bf load_frag16(const bf16_t* __restrict__ base,
                                             int ld, int k0, int lane) {
    const int m    = lane & 15;
    const int half = lane >> 4;
    const bf16_t* p = base + (size_t)m * ld + k0 + 8 * half;
    v16bf f;
#pragma unroll
    for (int e = 0; e < 8; ++e) f[e] = p[e];
#pragma unroll
    for (int e = 0; e < 8; ++e) f[e + 8] = p[e + 16];
    return f;
}

// ---- small utility kernels ------------------------------------------------

__global__ __launch_bounds__(256)
void cvt_kernel(const float* __restrict__ src, bf16_t* __restrict__ dst, int n) {
    for (int i = blockIdx.x * 256 + threadIdx.x; i < n; i += gridDim.x * 256)
        dst[i] = (bf16_t)src[i];
}

// Gather token embeddings -> bf16 activation matrices, row = t*16 + b.
__global__ __launch_bounds__(256)
void gather_kernel(const int* __restrict__ inputs, const int* __restrict__ outputs,
                   const float* __restrict__ emb,
                   bf16_t* __restrict__ encX, bf16_t* __restrict__ decX) {
    const int total = M_ROWS * DC;
    for (int idx = blockIdx.x * 256 + threadIdx.x; idx < total; idx += gridDim.x * 256) {
        const int c = idx & (DC - 1);
        const int r = idx >> 7;       // row = t*16 + b
        const int b = r & 15;
        const int t = r >> 4;
        encX[idx] = (bf16_t)emb[(size_t)inputs[b * T_SZ + t] * DC + c];
        decX[idx] = (t == 0) ? (bf16_t)0.0f
                             : (bf16_t)emb[(size_t)outputs[b * T_SZ + (t - 1)] * DC + c];
    }
}

// ---- input projection:  XP = X @ Wih^T + bih  (M=4096, K=128, N=512) ------

__global__ __launch_bounds__(256)
void input_proj_kernel(const bf16_t* __restrict__ X,    // [4096][128]
                       const bf16_t* __restrict__ Wih,  // [512][128] row-major = B(k,n)=Wih[n][k]
                       const float*  __restrict__ bih,  // [512]
                       float*        __restrict__ XP) { // [4096][512]
    __shared__ bf16_t Atile[16 * DC];                   // 4 KB
    const int tid = threadIdx.x;
    const int m0  = blockIdx.x * 16;
    {
        const unsigned* src = (const unsigned*)(X + (size_t)m0 * DC);
        unsigned*       dst = (unsigned*)Atile;
        for (int i = tid; i < 16 * DC / 2; i += 256) dst[i] = src[i];
    }
    __syncthreads();
    const int wave = tid >> 5, lane = tid & 31;
    const int n0   = (blockIdx.y * 8 + wave) * 16;
    v8f acc = {};
#pragma unroll
    for (int kt = 0; kt < DC / 32; ++kt) {
        v16bf a = load_frag16(Atile, DC, kt * 32, lane);
        v16bf b = load_frag16(Wih + (size_t)n0 * DC, DC, kt * 32, lane);
        acc = wmma_bf16(a, b, acc);
    }
    const int half = lane >> 4, n = n0 + (lane & 15);
    const float bn = bih[n];
#pragma unroll
    for (int r = 0; r < 8; ++r)
        XP[(size_t)(m0 + r + 8 * half) * DH + n] = acc[r] + bn;
}

// ---- persistent recurrence: h_{t+1} = tanh(xp_t + h_t @ Whh^T + bhh) ------
// 1 block, 512 threads = 16 waves; each wave owns two 16-wide N tiles.
// K-outer loop: one A (hidden state) fragment feeds both N tiles.

__global__ __launch_bounds__(512)
void rnn_kernel(const float*  __restrict__ xp,      // [T*16][512] (includes bih)
                const bf16_t* __restrict__ Whh,     // [512][512] bf16 (B(k,n)=Whh[n][k])
                const float*  __restrict__ bhh,     // [512]
                const bf16_t* __restrict__ h0,      // [16][512] or nullptr (zeros)
                bf16_t*       __restrict__ dstates, // [4096][512] row = b*T + t, or nullptr
                bf16_t*       __restrict__ hn_out)  // [16][512] or nullptr
{
    __shared__ bf16_t hbuf[2][B_SZ * DH];           // 2 x 16 KB double buffer
    const int tid  = threadIdx.x;
    const int wave = tid >> 5;
    const int lane = tid & 31;
    const int half = lane >> 4;

    {   // initialize h (bf16) in LDS
        unsigned* dst = (unsigned*)hbuf[0];
        if (h0) {
            const unsigned* src = (const unsigned*)h0;
            for (int i = tid; i < B_SZ * DH / 2; i += 512) dst[i] = src[i];
        } else {
            for (int i = tid; i < B_SZ * DH / 2; i += 512) dst[i] = 0u;
        }
    }
    __syncthreads();

    const int n0[2] = { wave * 32, wave * 32 + 16 };

    for (int t = 0; t < T_SZ; ++t) {
        const int cur = t & 1, nxt = cur ^ 1;
        v8f acc[2] = {};
#pragma unroll
        for (int kt = 0; kt < DH / 32; ++kt) {
            v16bf a = load_frag16(hbuf[cur], DH, kt * 32, lane);
#pragma unroll
            for (int ti = 0; ti < 2; ++ti) {
                v16bf b = load_frag16(Whh + (size_t)n0[ti] * DH, DH, kt * 32, lane);
                acc[ti] = wmma_bf16(a, b, acc[ti]);
            }
        }
#pragma unroll
        for (int ti = 0; ti < 2; ++ti) {
            const int n  = n0[ti] + (lane & 15);
            const float bn = bhh[n];
#pragma unroll
            for (int r = 0; r < 8; ++r) {
                const int m = r + 8 * half;                 // batch index
                float v = acc[ti][r] + xp[(size_t)(t * B_SZ + m) * DH + n] + bn;
                v = tanhf(v);
                const bf16_t vb = (bf16_t)v;
                hbuf[nxt][m * DH + n] = vb;
                if (dstates)
                    dstates[(size_t)(m * T_SZ + t) * DH + n] = vb;
            }
        }
        __syncthreads();
    }
    if (hn_out) {   // T=256 is even -> final state sits in hbuf[0]
        const unsigned* src = (const unsigned*)hbuf[0];
        unsigned*       dst = (unsigned*)hn_out;
        for (int i = tid; i < B_SZ * DH / 2; i += 512) dst[i] = src[i];
    }
}

// ---- output projection: logits = S @ out_W^T + out_b  (4096 x 32000, K=512)
// Block: 8 waves share a 16x512 A tile in LDS; each wave computes 16x80
// (5 N tiles), A fragment reused across the 5 WMMAs of each K step.

__global__ __launch_bounds__(256)
void out_gemm_kernel(const bf16_t* __restrict__ S,    // [4096][512]
                     const bf16_t* __restrict__ W,    // [32000][512]
                     const float*  __restrict__ bias, // [32000]
                     float*        __restrict__ out)  // [4096][32000]
{
    __shared__ bf16_t Atile[16 * DH];                 // 16 KB, shared by 8 waves
    const int tid = threadIdx.x;
    const int m0  = blockIdx.x * 16;
    {
        const unsigned* src = (const unsigned*)(S + (size_t)m0 * DH);
        unsigned*       dst = (unsigned*)Atile;
        for (int i = tid; i < 16 * DH / 2; i += 256) dst[i] = src[i];
    }
    const int wave = tid >> 5, lane = tid & 31;
    const int nbase = (blockIdx.y * 8 * NT_PER_WAVE + wave * NT_PER_WAVE) * 16;
    const bf16_t* Wbase = W + (size_t)nbase * DH;
#pragma unroll
    for (int nt = 0; nt < NT_PER_WAVE; ++nt)          // pull B rows toward L2/L0
        __builtin_prefetch(Wbase + (size_t)(nt * 16 + (lane & 15)) * DH, 0, 0);
    __syncthreads();

    v8f acc[NT_PER_WAVE] = {};
#pragma unroll
    for (int kt = 0; kt < DH / 32; ++kt) {
        v16bf a = load_frag16(Atile, DH, kt * 32, lane);
#pragma unroll
        for (int nt = 0; nt < NT_PER_WAVE; ++nt) {
            v16bf b = load_frag16(Wbase + (size_t)(nt * 16) * DH, DH, kt * 32, lane);
            acc[nt] = wmma_bf16(a, b, acc[nt]);
        }
    }
    const int half = lane >> 4;
#pragma unroll
    for (int nt = 0; nt < NT_PER_WAVE; ++nt) {
        const int n = nbase + nt * 16 + (lane & 15);
        const float bn = bias[n];
#pragma unroll
        for (int r = 0; r < 8; ++r)
            out[(size_t)(m0 + r + 8 * half) * V_SZ + n] = acc[nt][r] + bn;
    }
}

// ---- fused log-softmax: stage the whole 32000-float row in LDS ------------
// 125 KB dynamic LDS per block (CDNA5 WGP has 320 KB) -> one HBM read + one
// HBM write per logit instead of three reads.

__global__ __launch_bounds__(256)
void softmax_kernel(float* __restrict__ out) {
    extern __shared__ float srow[];                   // V_SZ floats (125 KB)
    __shared__ float red[256];
    const int r = blockIdx.x, tid = threadIdx.x;
    float* row = out + (size_t)r * V_SZ;

    float mx = -1e30f;
    for (int c = tid; c < V_SZ; c += 256) {
        const float v = row[c];
        srow[c] = v;
        mx = fmaxf(mx, v);
    }
    red[tid] = mx; __syncthreads();
    for (int s = 128; s > 0; s >>= 1) {
        if (tid < s) red[tid] = fmaxf(red[tid], red[tid + s]);
        __syncthreads();
    }
    const float rowmax = red[0]; __syncthreads();

    float sum = 0.f;
    for (int c = tid; c < V_SZ; c += 256) sum += expf(srow[c] - rowmax);
    red[tid] = sum; __syncthreads();
    for (int s = 128; s > 0; s >>= 1) {
        if (tid < s) red[tid] += red[tid + s];
        __syncthreads();
    }
    const float z = rowmax + logf(red[0]);
    __syncthreads();

    for (int c = tid; c < V_SZ; c += 256) row[c] = srow[c] - z;
}

// ---------------------------------------------------------------------------

extern "C" void kernel_launch(void* const* d_in, const int* in_sizes, int n_in,
                              void* d_out, int out_size, void* d_ws, size_t ws_size,
                              hipStream_t stream) {
    const int*   inputs  = (const int*)  d_in[0];
    const int*   outputs = (const int*)  d_in[1];
    const float* emb     = (const float*)d_in[2];
    const float* enc_Wih = (const float*)d_in[3];
    const float* enc_Whh = (const float*)d_in[4];
    const float* enc_bih = (const float*)d_in[5];
    const float* enc_bhh = (const float*)d_in[6];
    const float* dec_Wih = (const float*)d_in[7];
    const float* dec_Whh = (const float*)d_in[8];
    const float* dec_bih = (const float*)d_in[9];
    const float* dec_bhh = (const float*)d_in[10];
    const float* out_W   = (const float*)d_in[11];
    const float* out_b   = (const float*)d_in[12];
    float* out = (float*)d_out;

    // workspace carve-up (all offsets 256B aligned)
    char* ws = (char*)d_ws;
    size_t off = 0;
    bf16_t* encWhh_b = (bf16_t*)(ws + off); off += (size_t)DH * DH * 2;       // 512 KB
    bf16_t* decWhh_b = (bf16_t*)(ws + off); off += (size_t)DH * DH * 2;       // 512 KB
    bf16_t* encWih_b = (bf16_t*)(ws + off); off += (size_t)DH * DC * 2;       // 128 KB
    bf16_t* decWih_b = (bf16_t*)(ws + off); off += (size_t)DH * DC * 2;       // 128 KB
    bf16_t* outW_b   = (bf16_t*)(ws + off); off += (size_t)V_SZ * DH * 2;     // 32.8 MB
    bf16_t* encX     = (bf16_t*)(ws + off); off += (size_t)M_ROWS * DC * 2;   // 1 MB
    bf16_t* decX     = (bf16_t*)(ws + off); off += (size_t)M_ROWS * DC * 2;   // 1 MB
    float*  enc_xp   = (float*) (ws + off); off += (size_t)M_ROWS * DH * 4;   // 8 MB
    float*  dec_xp   = (float*) (ws + off); off += (size_t)M_ROWS * DH * 4;   // 8 MB
    bf16_t* dstates  = (bf16_t*)(ws + off); off += (size_t)M_ROWS * DH * 2;   // 4 MB
    bf16_t* hn_b     = (bf16_t*)(ws + off); off += (size_t)B_SZ * DH * 2;     // 16 KB
    (void)ws_size; (void)in_sizes; (void)n_in; (void)out_size;

    // 1) bf16 conversions (one-time per call; weights become L2-resident)
    cvt_kernel<<<1024, 256, 0, stream>>>(enc_Whh, encWhh_b, DH * DH);
    cvt_kernel<<<1024, 256, 0, stream>>>(dec_Whh, decWhh_b, DH * DH);
    cvt_kernel<<<256,  256, 0, stream>>>(enc_Wih, encWih_b, DH * DC);
    cvt_kernel<<<256,  256, 0, stream>>>(dec_Wih, decWih_b, DH * DC);
    cvt_kernel<<<4096, 256, 0, stream>>>(out_W,   outW_b,   V_SZ * DH);

    // 2) embedding gathers (+ decoder teacher-forcing shift)
    gather_kernel<<<1024, 256, 0, stream>>>(inputs, outputs, emb, encX, decX);

    // 3) input projections (WMMA): xp = x @ Wih^T + bih
    input_proj_kernel<<<dim3(M_ROWS / 16, 4), 256, 0, stream>>>(encX, encWih_b, enc_bih, enc_xp);
    input_proj_kernel<<<dim3(M_ROWS / 16, 4), 256, 0, stream>>>(decX, decWih_b, dec_bih, dec_xp);

    // 4) sequential recurrences (persistent single-workgroup WMMA kernels)
    rnn_kernel<<<1, 512, 0, stream>>>(enc_xp, encWhh_b, enc_bhh, nullptr, nullptr, hn_b);
    rnn_kernel<<<1, 512, 0, stream>>>(dec_xp, decWhh_b, dec_bhh, hn_b, dstates, nullptr);

    // 5) output projection (dominant GEMM, 134 GFLOP): logits -> d_out
    //    grid: 256 M tiles x 50 N groups (8 waves x 5 tiles x 16 cols = 640)
    out_gemm_kernel<<<dim3(M_ROWS / 16, V_SZ / (16 * 8 * NT_PER_WAVE)), 256, 0, stream>>>(
        dstates, outW_b, out_b, out);

    // 6) fused log-softmax (row staged in 125KB dynamic LDS)
    softmax_kernel<<<M_ROWS, 256, (size_t)V_SZ * sizeof(float), stream>>>(out);
}
